// Named_Entity_Recognition_11811160064592
// MI455X (gfx1250) — matrix operand order repaired and checked by reading, not compiled, not run on hardware
//
#include <hip/hip_runtime.h>
#include <hip/hip_bf16.h>
#include <stdint.h>

// ---------------- CDNA5 WMMA types ----------------
typedef __bf16 bf16_t;
typedef bf16_t v16bf __attribute__((ext_vector_type(16)));
typedef float  v8f   __attribute__((ext_vector_type(8)));
typedef unsigned int v4u __attribute__((ext_vector_type(4)));
typedef unsigned short u16;
typedef unsigned int   u32;

union BFrag { v16bf v; v4u u[2]; };

// Problem constants
#define BB 64
#define TT 512
#define FF 256
#define HH 256
#define H4 1024
#define KT 32
#define START_TAG 30
#define END_TAG 31

__device__ __forceinline__ u16 f2bf(float f) {
    u32 u = __float_as_uint(f);
    u32 r = u + 0x7fffu + ((u >> 16) & 1u);   // round-to-nearest-even
    return (u16)(r >> 16);
}
__device__ __forceinline__ float sigmoidf_(float x) {
    return 1.0f / (1.0f + __expf(-x));
}

// CDNA5 async global->LDS copy (16 bytes per lane), tracked by ASYNCcnt.
__device__ __forceinline__ void async_copy_b128(u32 lds_byte_off, const void* gptr) {
    asm volatile("global_load_async_to_lds_b128 %0, %1, off"
                 :: "v"(lds_byte_off), "v"(gptr) : "memory");
}
__device__ __forceinline__ void wait_asynccnt0() {
    asm volatile("s_wait_asynccnt 0" ::: "memory");
}

// ---------------- 1) f32 -> bf16 conversion ----------------
__global__ void f32_to_bf16_kernel(const float* __restrict__ in, u16* __restrict__ out, int n) {
    int i = blockIdx.x * blockDim.x + threadIdx.x;
    if (i < n) out[i] = f2bf(in[i]);
}

// ---------------- 2) embedding gather, (B,T) -> rows (t*64+b, F) bf16 ----------------
__global__ void gather_embed_kernel(const int* __restrict__ x, const float* __restrict__ emb,
                                    u16* __restrict__ e_bf) {
    int row = blockIdx.x;            // row = t*64 + b
    int t = row >> 6, b = row & 63;
    int v = x[b * TT + t];
    const float* src = emb + (size_t)v * FF;
    u16* dst = e_bf + (size_t)row * FF;
    for (int j = threadIdx.x; j < FF; j += blockDim.x) dst[j] = f2bf(src[j]);
}

// ---------------- 3) gx = e @ Wih^T + bias  (both directions) ----------------
// M = 32768 rows (t*64+b), N = 1024, K = 256.  One wave: 1 M-tile x 4 N-tiles.
// The 64 weight columns used by a block are block-uniform: stage them (32 KB)
// into LDS once via async global->LDS, then read B-fragments from LDS.
__global__ __launch_bounds__(256) void gx_gemm_kernel(
    const u16* __restrict__ e_bf,
    const u16* __restrict__ wih_f, const u16* __restrict__ wih_b,
    const float* __restrict__ bias_f, const float* __restrict__ bias_b,
    float* __restrict__ gx_f, float* __restrict__ gx_b) {

    __shared__ u16 lds_w[64][FF];     // 32 KB weight panel

    int tid  = threadIdx.x;
    int wid  = blockIdx.x * 8 + (tid >> 5);
    int lane = tid & 31;
    int grp = lane >> 4, ln = lane & 15;
    int mtile = wid & 2047;
    int rest  = blockIdx.x >> 8;      // block-uniform (2048 % 8 == 0)
    int dir   = rest >> 4;
    int ngrp  = rest & 15;            // 4 N-tiles each
    const u16*  w    = dir ? wih_b  : wih_f;
    const float* bia = dir ? bias_b : bias_f;
    float* gx        = dir ? gx_b   : gx_f;

    // async-stage the 64x256 bf16 weight panel into LDS
    {
        const u16* wbase = w + (size_t)(ngrp * 64) * FF;
        u32 lds_base = (u32)(uintptr_t)&lds_w[0][0];
        for (int i = tid; i < 2048; i += 256)          // 2048 x 16B = 32 KB
            async_copy_b128(lds_base + i * 16, wbase + i * 8);
        wait_asynccnt0();
        __syncthreads();
    }

    v8f acc[4] = {v8f{0}, v8f{0}, v8f{0}, v8f{0}};
    int arow = mtile * 16 + ln;
#pragma unroll
    for (int k = 0; k < 8; ++k) {
        BFrag A;
        const u16* ap = e_bf + (size_t)arow * FF + k * 32 + grp * 8;
        A.u[0] = *(const v4u*)ap;
        A.u[1] = *(const v4u*)(ap + 16);
#pragma unroll
        for (int j = 0; j < 4; ++j) {
            const u16* bp = &lds_w[j * 16 + ln][k * 32 + grp * 8];
            BFrag Bf;
            Bf.u[0] = *(const v4u*)bp;
            Bf.u[1] = *(const v4u*)(bp + 16);
            acc[j] = __builtin_amdgcn_wmma_f32_16x16x32_bf16(
                false, A.v, false, Bf.v, (short)0, acc[j], false, false);
        }
    }
#pragma unroll
    for (int j = 0; j < 4; ++j) {
        int n = (ngrp * 4 + j) * 16 + ln;
        float bb = bia[n];
#pragma unroll
        for (int v = 0; v < 8; ++v) {
            int r = mtile * 16 + grp * 8 + v;
            gx[(size_t)r * H4 + n] = acc[j][v] + bb;
        }
    }
}

// ---------------- 4) sequential LSTM scan (one block per direction) ----------------
// 32 waves: wave = mtile(4) x hgroup(8). Each wave: 2 h-tiles x 4 gate-tiles.
// h double-buffered bf16 in LDS (2 x 64x256), c in registers. Whh (512 KB bf16)
// exceeds LDS, so it streams from L2 each step (2 MB/step, trivial at L2 BW).
__global__ __launch_bounds__(1024) void lstm_scan_kernel(
    const float* __restrict__ gx_f, const float* __restrict__ gx_b,
    const u16* __restrict__ whh_f, const u16* __restrict__ whh_b,
    u16* __restrict__ h_cat) {

    __shared__ u16 lds_h[2][BB][HH];   // 64 KB

    int dir = blockIdx.x;
    const float* gx  = dir ? gx_b  : gx_f;
    const u16*   whh = dir ? whh_b : whh_f;

    int tid = threadIdx.x;
    int lane = tid & 31, wave = tid >> 5;
    int grp = lane >> 4, ln = lane & 15;
    int mtile = wave & 3, hgroup = wave >> 2;

    // zero initial h buffer
    for (int i = tid; i < BB * HH; i += 1024) ((u16*)lds_h)[i] = 0;
    __syncthreads();

    float creg[2][8];
#pragma unroll
    for (int a = 0; a < 2; ++a)
#pragma unroll
        for (int v = 0; v < 8; ++v) creg[a][v] = 0.0f;

    int p = 0;
    for (int t = 0; t < TT; ++t) {
        int te = dir ? (TT - 1 - t) : t;
        const float* gxt = gx + (size_t)te * BB * H4;
        // prefetch next timestep's gx slice (global_prefetch path)
        if (t + 1 < TT) {
            int tn = dir ? (te - 1) : (te + 1);
            __builtin_prefetch(gx + (size_t)tn * BB * H4 + tid * 16, 0, 1);
        }

        // preload accumulators with gx (C layout: M = v + 8*grp, N = ln)
        v8f acc[2][4];
#pragma unroll
        for (int ht = 0; ht < 2; ++ht) {
            int hc = (hgroup * 2 + ht) * 16 + ln;
#pragma unroll
            for (int g = 0; g < 4; ++g) {
                int n = g * HH + hc;
#pragma unroll
                for (int v = 0; v < 8; ++v) {
                    int b = mtile * 16 + grp * 8 + v;
                    acc[ht][g][v] = gxt[(size_t)b * H4 + n];
                }
            }
        }

        // z += h_prev @ Whh^T  via 64 WMMAs per wave
#pragma unroll
        for (int k = 0; k < 8; ++k) {
            BFrag A;
            const u16* ap = &lds_h[p][mtile * 16 + ln][k * 32 + grp * 8];
            A.u[0] = *(const v4u*)ap;
            A.u[1] = *(const v4u*)(ap + 16);
#pragma unroll
            for (int ht = 0; ht < 2; ++ht) {
#pragma unroll
                for (int g = 0; g < 4; ++g) {
                    int n = g * HH + (hgroup * 2 + ht) * 16 + ln;
                    const u16* bp = whh + (size_t)n * HH + k * 32 + grp * 8;
                    BFrag Bf;
                    Bf.u[0] = *(const v4u*)bp;
                    Bf.u[1] = *(const v4u*)(bp + 16);
                    acc[ht][g] = __builtin_amdgcn_wmma_f32_16x16x32_bf16(
                        false, A.v, false, Bf.v, (short)0, acc[ht][g], false, false);
                }
            }
        }

        // gates: i,f,g,o for same (b,h) live in same lane/slot across the 4 gate tiles
#pragma unroll
        for (int ht = 0; ht < 2; ++ht) {
            int hcol = (hgroup * 2 + ht) * 16 + ln;
#pragma unroll
            for (int v = 0; v < 8; ++v) {
                int b = mtile * 16 + grp * 8 + v;
                float ig = sigmoidf_(acc[ht][0][v]);
                float fg = sigmoidf_(acc[ht][1][v]);
                float gg = tanhf(acc[ht][2][v]);
                float og = sigmoidf_(acc[ht][3][v]);
                float cn = fg * creg[ht][v] + ig * gg;
                float hn = og * tanhf(cn);
                creg[ht][v] = cn;
                u16 hb = f2bf(hn);
                lds_h[p ^ 1][b][hcol] = hb;
                // concatenated layout (t*64+b, 2H): forward in [0,256), backward in [256,512)
                h_cat[((size_t)te * BB + b) * (2 * HH) + dir * HH + hcol] = hb;
            }
        }
        __syncthreads();
        p ^= 1;
    }
}

// ---------------- 5) scores = h_cat @ Wfc^T + b_fc ----------------
// M = 32768, N = 32 (2 tiles), K = 512 (16 steps). One wave per M-tile.
// Whole Wfc (32 KB bf16) async-staged into LDS once per block.
__global__ __launch_bounds__(256) void scores_gemm_kernel(
    const u16* __restrict__ h_cat, const u16* __restrict__ wfc_bf,
    const float* __restrict__ b_fc, float* __restrict__ scores) {

    __shared__ u16 lds_w[KT][2 * HH];  // 32 x 512 bf16 = 32 KB

    int tid = threadIdx.x;
    {
        u32 lds_base = (u32)(uintptr_t)&lds_w[0][0];
        for (int i = tid; i < 2048; i += 256)          // 2048 x 16B = 32 KB
            async_copy_b128(lds_base + i * 16, wfc_bf + i * 8);
        wait_asynccnt0();
        __syncthreads();
    }

    int mtile = blockIdx.x * 8 + (tid >> 5);
    int lane = tid & 31;
    int grp = lane >> 4, ln = lane & 15;

    v8f acc[2] = {v8f{0}, v8f{0}};
    int arow = mtile * 16 + ln;
#pragma unroll
    for (int k = 0; k < 16; ++k) {
        BFrag A;
        const u16* ap = h_cat + (size_t)arow * (2 * HH) + k * 32 + grp * 8;
        A.u[0] = *(const v4u*)ap;
        A.u[1] = *(const v4u*)(ap + 16);
#pragma unroll
        for (int nt = 0; nt < 2; ++nt) {
            const u16* bp = &lds_w[nt * 16 + ln][k * 32 + grp * 8];
            BFrag Bf;
            Bf.u[0] = *(const v4u*)bp;
            Bf.u[1] = *(const v4u*)(bp + 16);
            acc[nt] = __builtin_amdgcn_wmma_f32_16x16x32_bf16(
                false, A.v, false, Bf.v, (short)0, acc[nt], false, false);
        }
    }
#pragma unroll
    for (int nt = 0; nt < 2; ++nt) {
        int n = nt * 16 + ln;
        float bb = b_fc[n];
#pragma unroll
        for (int v = 0; v < 8; ++v) {
            int r = mtile * 16 + grp * 8 + v;
            scores[(size_t)r * KT + n] = acc[nt][v] + bb;
        }
    }
}

// ---------------- 6) CRF loss: wave per batch, lane = tag ----------------
__global__ __launch_bounds__(1024) void crf_kernel(
    const float* __restrict__ scores, const int* __restrict__ tags,
    const int* __restrict__ mask, const float* __restrict__ transition,
    float* __restrict__ loss_part) {
    int b = blockIdx.x * 32 + (threadIdx.x >> 5);
    int j = threadIdx.x & 31;

    float tr[KT];
#pragma unroll
    for (int i = 0; i < KT; ++i) tr[i] = transition[i * KT + j];

    float alpha = transition[START_TAG * KT + j] + scores[(size_t)(0 * BB + b) * KT + j];

    for (int t = 1; t < TT; ++t) {
        int m = mask[b * TT + t];
        float sc = scores[((size_t)t * BB + b) * KT + j];
        float vv[KT];
        float vmax = -1e30f;
#pragma unroll
        for (int i = 0; i < KT; ++i) {
            vv[i] = __shfl(alpha, i) + tr[i];
            vmax = fmaxf(vmax, vv[i]);
        }
        float s = 0.0f;
#pragma unroll
        for (int i = 0; i < KT; ++i) s += __expf(vv[i] - vmax);
        float na = vmax + __logf(s) + sc;
        alpha = (m > 0) ? na : alpha;
    }

    // total_prob = LSE_j(alpha_j + transition[j][END])
    float fin = alpha + transition[j * KT + END_TAG];
    float m = fin;
    for (int off = 16; off > 0; off >>= 1) m = fmaxf(m, __shfl_xor(m, off));
    float e = __expf(fin - m);
    for (int off = 16; off > 0; off >>= 1) e += __shfl_xor(e, off);
    float total_prob = m + __logf(e);

    // gold-path score, distributed over lanes
    float body = 0.0f;
    int cnt = 0;
    for (int t = j; t < TT; t += 32) {
        int mk = mask[b * TT + t];
        cnt += mk;
        if (t > 0 && mk) {
            int tg = tags[b * TT + t];
            int tp = tags[b * TT + t - 1];
            body += transition[tp * KT + tg] + scores[((size_t)t * BB + b) * KT + tg];
        }
    }
    for (int off = 16; off > 0; off >>= 1) {
        body += __shfl_xor(body, off);
        cnt  += __shfl_xor(cnt, off);
    }
    int last_tag = tags[b * TT + (cnt - 1)];
    int t0 = tags[b * TT + 0];
    float first = transition[START_TAG * KT + t0] + scores[(size_t)b * KT + t0];
    float true_prob = first + body + transition[last_tag * KT + END_TAG];
    if (j == 0) loss_part[b] = -(true_prob - total_prob);
}

// ---------------- 7) final reduction ----------------
__global__ void final_reduce_kernel(const float* __restrict__ loss_part, float* __restrict__ out) {
    int j = threadIdx.x;  // 32 threads
    float v = loss_part[j] + loss_part[j + 32];
    for (int off = 16; off > 0; off >>= 1) v += __shfl_xor(v, off);
    if (j == 0) out[0] = v;
}

// ---------------- host launcher ----------------
extern "C" void kernel_launch(void* const* d_in, const int* in_sizes, int n_in,
                              void* d_out, int out_size, void* d_ws, size_t ws_size,
                              hipStream_t stream) {
    const int*   x      = (const int*)d_in[0];
    const int*   tags   = (const int*)d_in[1];
    const int*   mask   = (const int*)d_in[2];
    const float* emb    = (const float*)d_in[3];
    const float* w_ih_f = (const float*)d_in[4];
    const float* w_hh_f = (const float*)d_in[5];
    const float* b_f    = (const float*)d_in[6];
    const float* w_ih_b = (const float*)d_in[7];
    const float* w_hh_b = (const float*)d_in[8];
    const float* b_b    = (const float*)d_in[9];
    const float* w_fc   = (const float*)d_in[10];
    const float* b_fc   = (const float*)d_in[11];
    const float* trans  = (const float*)d_in[12];

    char* ws = (char*)d_ws;
    const size_t SZ_WIH = (size_t)H4 * FF * 2;      // 512 KB
    const size_t SZ_WFC = (size_t)KT * (2 * HH) * 2;
    u16* wihf_bf = (u16*)(ws); ws += SZ_WIH;
    u16* whhf_bf = (u16*)(ws); ws += SZ_WIH;
    u16* wihb_bf = (u16*)(ws); ws += SZ_WIH;
    u16* whhb_bf = (u16*)(ws); ws += SZ_WIH;
    u16* wfc_bf  = (u16*)(ws); ws += ((SZ_WFC + 511) & ~(size_t)511);
    u16* e_bf    = (u16*)(ws); ws += (size_t)TT * BB * FF * 2;         // 16 MB
    float* gxf   = (float*)(ws); ws += (size_t)TT * BB * H4 * 4;       // 128 MB
    float* gxb   = (float*)(ws); ws += (size_t)TT * BB * H4 * 4;       // 128 MB
    u16* h_cat   = (u16*)(ws); ws += (size_t)TT * BB * (2 * HH) * 2;   // 32 MB
    float* scores = (float*)(ws); ws += (size_t)TT * BB * KT * 4;      // 4 MB
    float* loss_part = (float*)(ws); ws += 256;

    // 1) weight conversion
    f32_to_bf16_kernel<<<(H4 * FF + 255) / 256, 256, 0, stream>>>(w_ih_f, wihf_bf, H4 * FF);
    f32_to_bf16_kernel<<<(H4 * HH + 255) / 256, 256, 0, stream>>>(w_hh_f, whhf_bf, H4 * HH);
    f32_to_bf16_kernel<<<(H4 * FF + 255) / 256, 256, 0, stream>>>(w_ih_b, wihb_bf, H4 * FF);
    f32_to_bf16_kernel<<<(H4 * HH + 255) / 256, 256, 0, stream>>>(w_hh_b, whhb_bf, H4 * HH);
    f32_to_bf16_kernel<<<(KT * 2 * HH + 255) / 256, 256, 0, stream>>>(w_fc, wfc_bf, KT * 2 * HH);

    // 2) embedding gather into (t*64+b, F) bf16
    gather_embed_kernel<<<TT * BB, 256, 0, stream>>>(x, emb, e_bf);

    // 3) input GEMMs, both directions: 65536 waves
    gx_gemm_kernel<<<8192, 256, 0, stream>>>(e_bf, wihf_bf, wihb_bf, b_f, b_b, gxf, gxb);

    // 4) sequential bidirectional LSTM scan (block 0 = fwd, block 1 = bwd)
    lstm_scan_kernel<<<2, 1024, 0, stream>>>(gxf, gxb, whhf_bf, whhb_bf, h_cat);

    // 5) emission scores
    scores_gemm_kernel<<<256, 256, 0, stream>>>(h_cat, wfc_bf, b_fc, scores);

    // 6) CRF forward + gold path (64 waves = 64 batches)
    crf_kernel<<<2, 1024, 0, stream>>>(scores, tags, mask, trans, loss_part);

    // 7) reduce to scalar loss
    final_reduce_kernel<<<1, 32, 0, stream>>>(loss_part, (float*)d_out);
}
